// SelfAttention_14817637171648
// MI455X (gfx1250) — matrix operand order repaired and checked by reading, not compiled, not run on hardware
//
#include <hip/hip_runtime.h>
#include <stdint.h>

#define SEQ   3072
#define DIM   1536
#define HEADS 12
#define HD    128
#define EPSF  1e-6f

typedef __bf16 v16bf __attribute__((ext_vector_type(16)));
typedef float  v8f   __attribute__((ext_vector_type(8)));

union FragU { uint32_t u[8]; v16bf v; };

__device__ __forceinline__ uint16_t f2bf(float f) {
    uint32_t u = __float_as_uint(f);
    u += 0x7fffu + ((u >> 16) & 1u);           // round-to-nearest-even
    return (uint16_t)(u >> 16);
}
__device__ __forceinline__ uint32_t pk2(float a, float b) {
    return (uint32_t)f2bf(a) | ((uint32_t)f2bf(b) << 16);
}
__device__ __forceinline__ float bf_lo(uint32_t w) { return __uint_as_float(w << 16); }
__device__ __forceinline__ float bf_hi(uint32_t w) { return __uint_as_float(w & 0xffff0000u); }

__device__ __forceinline__ v8f wmma_bf16(v16bf a, v16bf b, v8f c) {
    return __builtin_amdgcn_wmma_f32_16x16x32_bf16(false, a, false, b,
                                                   (short)0, c, false, false);
}

// ---- fragment loaders (contiguous k order: k = 16*h + s); tiles are bf16 packed in u32 LDS ----
__device__ __forceinline__ v16bf load_a(const uint32_t* t, int stride, int m0) {
    const int lane = threadIdx.x & 31;
    const int m = m0 + (lane & 15), h = lane >> 4;
    FragU f;
#pragma unroll
    for (int r = 0; r < 8; ++r) {
        int k = (h << 4) + (r << 1);
        f.u[r] = t[m * stride + (k >> 1)];
    }
    return f.v;
}
__device__ __forceinline__ v16bf load_bk(const uint32_t* t, int stride, int n0) {
    const int lane = threadIdx.x & 31;
    const int n = n0 + (lane & 15), h = lane >> 4;
    FragU f;
#pragma unroll
    for (int r = 0; r < 8; ++r) {
        int k = (h << 4) + (r << 1);
        f.u[r] = t[n * stride + (k >> 1)];
    }
    return f.v;
}

// ---- CDNA5 async global->LDS (ASYNCcnt-tracked) ----
__device__ __forceinline__ void async_b128(uint32_t lds_off, const void* base, uint32_t goff) {
    asm volatile("global_load_async_to_lds_b128 %0, %1, %2 offset:0"
                 :: "v"(lds_off), "v"(goff), "s"(base) : "memory");
}
template <int N>
__device__ __forceinline__ void wait_async_le() {
    asm volatile("s_wait_asynccnt %0" :: "i"(N) : "memory");
}

// =====================  fp32 -> bf16 pack (pre-convert kernel)  =====================
__global__ __launch_bounds__(256)
void cvt_bf16(const float* __restrict__ src, uint32_t* __restrict__ dst, int n2)
{
    int i = blockIdx.x * 256 + threadIdx.x;
    if (i < n2) {
        float2 f = ((const float2*)src)[i];
        dst[i] = pk2(f.x, f.y);
    }
}

// =====================  GEMM: Out[M,N] = A[M,K] * W[N,K]^T + bias  =====================
// bf16 operands; block tile 128x128, K-step 32, double-buffered async pipeline.
// 8 waves -> 32x64 tile each: 2 m-subtiles x 4 n-subtiles = 8 wmma per stage.
__device__ __forceinline__ void gemm_issue(const uint16_t* A, const uint16_t* W,
                                           uint32_t* As, uint32_t* Bs,
                                           int m0, int n0, int K, int k0, int tid)
{
#pragma unroll
    for (int i = 0; i < 2; ++i) {
        int c = tid + (i << 8);            // 0..511
        int r = c >> 2, cc = c & 3;        // row 0..127, b128-chunk 0..3
        uint32_t loff = (uint32_t)(r * 20 + (cc << 2)) * 4u;
        async_b128((uint32_t)(uintptr_t)As + loff, A,
                   (uint32_t)((m0 + r) * (K * 2) + k0 * 2 + (cc << 4)));
        async_b128((uint32_t)(uintptr_t)Bs + loff, W,
                   (uint32_t)((n0 + r) * (K * 2) + k0 * 2 + (cc << 4)));
    }
}

template <bool OUT_BF16>
__global__ __launch_bounds__(256)
void gemm_bf16(const uint16_t* __restrict__ A, const uint16_t* __restrict__ W,
               const float* __restrict__ bias, void* __restrict__ Outp,
               int M, int N, int K)
{
    __shared__ uint32_t As[2][128 * 20];   // 128 x 32 bf16, stride 80B (16B-aligned)
    __shared__ uint32_t Bs[2][128 * 20];

    const int tid  = threadIdx.x;
    const int lane = tid & 31;
    const int wv   = tid >> 5;
    const int nt   = N >> 7;
    const int m0   = (blockIdx.x / nt) << 7;
    const int n0   = (blockIdx.x % nt) << 7;
    const int wm   = (wv & 3) << 5;     // 0,32,64,96
    const int wn   = (wv >> 2) << 6;    // 0,64

    v8f acc[2][4] = {};

    gemm_issue(A, W, As[0], Bs[0], m0, n0, K, 0, tid);   // prologue stage 0
    int buf = 0;
    for (int k0 = 0; k0 < K; k0 += 32) {
        const bool more = (k0 + 32) < K;
        if (more) {
            gemm_issue(A, W, As[buf ^ 1], Bs[buf ^ 1], m0, n0, K, k0 + 32, tid);
            __builtin_prefetch(A + (size_t)(m0 + (tid >> 1)) * K + k0 + 64, 0, 1);
            __builtin_prefetch(W + (size_t)(n0 + (tid >> 1)) * K + k0 + 64, 0, 1);
            wait_async_le<4>();          // stage k0's 4 copies landed (in-order)
        } else {
            wait_async_le<0>();
        }
        __syncthreads();

        v16bf a0 = load_a(As[buf], 20, wm);
        v16bf a1 = load_a(As[buf], 20, wm + 16);
#pragma unroll
        for (int ni = 0; ni < 4; ++ni) {
            v16bf b = load_bk(Bs[buf], 20, wn + (ni << 4));
            acc[0][ni] = wmma_bf16(a0, b, acc[0][ni]);
            acc[1][ni] = wmma_bf16(a1, b, acc[1][ni]);
        }
        __syncthreads();
        buf ^= 1;
    }

    const int half = lane >> 4, ln = lane & 15;
    float*    Outf = (float*)Outp;
    uint16_t* Outh = (uint16_t*)Outp;
#pragma unroll
    for (int mi = 0; mi < 2; ++mi)
#pragma unroll
        for (int ni = 0; ni < 4; ++ni) {
            int n = n0 + wn + (ni << 4) + ln;
            float bz = bias[n];
#pragma unroll
            for (int r = 0; r < 8; ++r) {
                int m = m0 + wm + (mi << 4) + (half << 3) + r;
                float v = acc[mi][ni][r] + bz;
                if (OUT_BF16) Outh[(size_t)m * N + n] = f2bf(v);
                else          Outf[(size_t)m * N + n] = v;
            }
        }
}

// =====================  RMSNorm + RoPE + layout (reads bf16 raw QKV)  =====================
__global__ __launch_bounds__(256)
void norm_rope(const uint16_t* __restrict__ qr, const uint16_t* __restrict__ kr,
               const uint16_t* __restrict__ vr, const float* __restrict__ gq,
               const float* __restrict__ gk, const float* __restrict__ fre,
               uint32_t* __restrict__ qb, uint32_t* __restrict__ kb,
               uint16_t* __restrict__ vbT)
{
    const int lane = threadIdx.x & 31;
    const int row  = blockIdx.x * 8 + (threadIdx.x >> 5);
    const int kind = row / SEQ;
    const int s    = row - kind * SEQ;

    if (kind == 2) {   // V: transpose to [DIM][SEQ]
        const uint32_t* src = (const uint32_t*)vr + (size_t)s * (DIM / 2);
#pragma unroll
        for (int i = 0; i < 24; ++i) {
            int pp = lane + (i << 5);
            uint32_t w = src[pp];
            vbT[(size_t)(pp << 1) * SEQ + s]       = (uint16_t)(w & 0xffffu);
            vbT[(size_t)((pp << 1) + 1) * SEQ + s] = (uint16_t)(w >> 16);
        }
        return;
    }
    const uint32_t* src = (const uint32_t*)(kind == 0 ? qr : kr) + (size_t)s * (DIM / 2);
    const float*    g   = kind == 0 ? gq : gk;
    uint32_t*       dst = (kind == 0 ? qb : kb) + (size_t)s * (DIM / 2);

    float ss = 0.f;
    uint32_t wv_[24];
#pragma unroll
    for (int i = 0; i < 24; ++i) {
        int pp = lane + (i << 5);
        uint32_t w = src[pp];
        wv_[i] = w;
        float a = bf_lo(w), b = bf_hi(w);
        ss += a * a + b * b;
    }
#pragma unroll
    for (int m = 1; m < 32; m <<= 1) ss += __shfl_xor(ss, m, 32);
    const float scale = rsqrtf(ss * (1.0f / DIM) + EPSF);
#pragma unroll
    for (int i = 0; i < 24; ++i) {
        int pp = lane + (i << 5);
        uint32_t w = wv_[i];
        float a = bf_lo(w) * scale * g[pp << 1];
        float b = bf_hi(w) * scale * g[(pp << 1) + 1];
        int p = pp & 63;
        float f0 = fre[s * 128 + (p << 1)];
        float f1 = fre[s * 128 + (p << 1) + 1];
        dst[pp] = pk2(a * f0 - b * f1, a * f1 + b * f0);
    }
}

// =====================  Flash attention (pipelined async KV, bf16 WMMA)  =====================
// grid = HEADS*(SEQ/128); 8 waves, wave owns 16 q rows. Q-stage LDS is reused as
// double-buffered K / V-transposed tiles once Q fragments are in registers.
// smem word map:  phase1: [0, 8704) = Q tile (128x128 bf16, stride 68)
//                 phase2: [0,2176)=K buf0, [2176,4352)=K buf1 (32x128, stride 68)
//                         [4352,6912)=Vt buf0, [6912,9472)=Vt buf1 (128x32, stride 20)
//                 always: [9472,11648) = P tile (128x32 bf16, stride 17)
#define KS_W   2176
#define VT_OFF 4352
#define VT_W   2560
#define PS_OFF 9472
__global__ __launch_bounds__(256)
void flash_attn(const uint16_t* __restrict__ qb, const uint16_t* __restrict__ kb,
                const uint16_t* __restrict__ vbT, uint16_t* __restrict__ ob)
{
    __shared__ uint32_t smem[PS_OFF + 2176];

    const int tid  = threadIdx.x;
    const int lane = tid & 31;
    const int wv   = tid >> 5;
    const int head = blockIdx.x / (SEQ / 128);
    const int q0   = (blockIdx.x % (SEQ / 128)) << 7;
    const int wm   = wv << 4;
    const int half = lane >> 4, ln = lane & 15;
    const uint16_t* vh = vbT + (size_t)head * HD * SEQ;

    // phase 1: async-load Q tile (128 rows x 16 b128-chunks), build per-wave A-frags
#pragma unroll
    for (int i = 0; i < 8; ++i) {
        int c = tid + (i << 8);
        int r = c >> 4, cc = c & 15;
        uint32_t goff = (uint32_t)((q0 + r) * (DIM * 2) + head * (HD * 2) + (cc << 4));
        async_b128((uint32_t)(uintptr_t)&smem[r * 68 + (cc << 2)], qb, goff);
    }
    wait_async_le<0>();
    __syncthreads();
    v16bf qa[4];
#pragma unroll
    for (int kc = 0; kc < 4; ++kc) qa[kc] = load_a(smem + (kc << 4), 68, wm);
    __syncthreads();                      // Q region may now be recycled

    // KV tile issue: K 512 chunks + Vt 512 chunks -> 4 async per thread
    auto issue_kv = [&](int buf, int kv0) {
        uint32_t ksb = (uint32_t)(uintptr_t)smem + (uint32_t)(buf * KS_W) * 4u;
        uint32_t vtb = (uint32_t)(uintptr_t)smem + (uint32_t)(VT_OFF + buf * VT_W) * 4u;
#pragma unroll
        for (int i = 0; i < 2; ++i) {
            int c  = tid + (i << 8);
            int kr = c >> 4, kc = c & 15;
            uint32_t gk = (uint32_t)((kv0 + kr) * (DIM * 2) + head * (HD * 2) + (kc << 4));
            async_b128(ksb + (uint32_t)(kr * 68 + (kc << 2)) * 4u, kb, gk);
            int vr = c >> 2, vc = c & 3;
            uint32_t gv = (uint32_t)(vr * (SEQ * 2) + kv0 * 2 + (vc << 4));
            async_b128(vtb + (uint32_t)(vr * 20 + (vc << 2)) * 4u, vh, gv);
        }
    };

    v8f o[8] = {};
    float mrow[8], lrow[8];
#pragma unroll
    for (int r = 0; r < 8; ++r) { mrow[r] = -3.0e38f; lrow[r] = 0.f; }
    const float sscale = 0.08838834764831845f;   // 1/sqrt(128)

    issue_kv(0, 0);                       // prologue
    int buf = 0;
    for (int kv0 = 0; kv0 < SEQ; kv0 += 32) {
        const bool more = (kv0 + 32) < SEQ;
        if (more) { issue_kv(buf ^ 1, kv0 + 32); wait_async_le<4>(); }
        else      { wait_async_le<0>(); }
        __syncthreads();
        const uint32_t* Ks = smem + buf * KS_W;
        const uint32_t* Vt = smem + VT_OFF + buf * VT_W;

        // scores S = Q K^T : 16 x 32 per wave
        v8f sc[2] = {};
#pragma unroll
        for (int kc = 0; kc < 4; ++kc) {
            v16bf b0 = load_bk(Ks + (kc << 4), 68, 0);
            v16bf b1 = load_bk(Ks + (kc << 4), 68, 16);
            sc[0] = wmma_bf16(qa[kc], b0, sc[0]);
            sc[1] = wmma_bf16(qa[kc], b1, sc[1]);
        }

        // online softmax; rows are wm + 8*half + r, cols across 16 lanes
#pragma unroll
        for (int r = 0; r < 8; ++r) {
            float s0 = sc[0][r] * sscale;
            float s1 = sc[1][r] * sscale;
            float mx = fmaxf(s0, s1);
#pragma unroll
            for (int m = 1; m < 16; m <<= 1) mx = fmaxf(mx, __shfl_xor(mx, m, 32));
            float mnew  = fmaxf(mrow[r], mx);
            float alpha = __expf(mrow[r] - mnew);
            float p0 = __expf(s0 - mnew);
            float p1 = __expf(s1 - mnew);
            float ps = p0 + p1;
#pragma unroll
            for (int m = 1; m < 16; m <<= 1) ps += __shfl_xor(ps, m, 32);
            lrow[r] = lrow[r] * alpha + ps;
            mrow[r] = mnew;
            int prow = wm + (half << 3) + r;
            uint16_t* p16 = (uint16_t*)&smem[PS_OFF + prow * 17];
            p16[ln]      = f2bf(p0);
            p16[16 + ln] = f2bf(p1);
#pragma unroll
            for (int sub = 0; sub < 8; ++sub) o[sub][r] *= alpha;
        }

        // O += P V : P A-fragment (own rows); V from k-major transposed tile
        v16bf pa = load_a(smem + PS_OFF, 17, wm);
#pragma unroll
        for (int sub = 0; sub < 8; ++sub) {
            v16bf bvf = load_bk(Vt, 20, sub << 4);
            o[sub] = wmma_bf16(pa, bvf, o[sub]);
        }
        __syncthreads();
        buf ^= 1;
    }

    // normalize and store bf16 attention output [s][DIM]
#pragma unroll
    for (int r = 0; r < 8; ++r) {
        float inv = 1.0f / lrow[r];
        int srow = q0 + wm + (half << 3) + r;
#pragma unroll
        for (int sub = 0; sub < 8; ++sub) {
            ob[(size_t)srow * DIM + head * HD + (sub << 4) + ln] = f2bf(o[sub][r] * inv);
        }
    }
}

// =====================  launcher  =====================
extern "C" void kernel_launch(void* const* d_in, const int* in_sizes, int n_in,
                              void* d_out, int out_size, void* d_ws, size_t ws_size,
                              hipStream_t stream)
{
    (void)in_sizes; (void)n_in; (void)out_size; (void)ws_size;
    const float* x  = (const float*)d_in[0];
    const float* fq = (const float*)d_in[1];
    const float* wq = (const float*)d_in[2];
    const float* bq = (const float*)d_in[3];
    const float* wk = (const float*)d_in[4];
    const float* bk = (const float*)d_in[5];
    const float* wv = (const float*)d_in[6];
    const float* bv = (const float*)d_in[7];
    const float* wo = (const float*)d_in[8];
    const float* bo = (const float*)d_in[9];
    const float* gq = (const float*)d_in[10];
    const float* gk = (const float*)d_in[11];

    char* ws = (char*)d_ws;
    const size_t hb = (size_t)SEQ * DIM * 2;     // bf16 activation plane
    const size_t wb = (size_t)DIM * DIM * 2;     // bf16 weight plane
    uint16_t* xb   = (uint16_t*)(ws);
    uint16_t* wqb  = (uint16_t*)(ws + hb);
    uint16_t* wkb  = (uint16_t*)(ws + hb + wb);
    uint16_t* wvb  = (uint16_t*)(ws + hb + 2 * wb);
    uint16_t* wob  = (uint16_t*)(ws + hb + 3 * wb);
    uint16_t* qraw = (uint16_t*)(ws + hb + 4 * wb);
    uint16_t* kraw = (uint16_t*)(ws + 2 * hb + 4 * wb);
    uint16_t* vraw = (uint16_t*)(ws + 3 * hb + 4 * wb);
    uint16_t* qbuf = (uint16_t*)(ws + 4 * hb + 4 * wb);
    uint16_t* kbuf = (uint16_t*)(ws + 5 * hb + 4 * wb);
    uint16_t* vbT  = (uint16_t*)(ws + 6 * hb + 4 * wb);
    uint16_t* obuf = (uint16_t*)(ws + 7 * hb + 4 * wb);

    dim3 blk(256);
    const int xn2 = SEQ * DIM / 2, wn2 = DIM * DIM / 2;
    cvt_bf16<<<dim3((xn2 + 255) / 256), blk, 0, stream>>>(x,  (uint32_t*)xb,  xn2);
    cvt_bf16<<<dim3((wn2 + 255) / 256), blk, 0, stream>>>(wq, (uint32_t*)wqb, wn2);
    cvt_bf16<<<dim3((wn2 + 255) / 256), blk, 0, stream>>>(wk, (uint32_t*)wkb, wn2);
    cvt_bf16<<<dim3((wn2 + 255) / 256), blk, 0, stream>>>(wv, (uint32_t*)wvb, wn2);
    cvt_bf16<<<dim3((wn2 + 255) / 256), blk, 0, stream>>>(wo, (uint32_t*)wob, wn2);

    dim3 gg((SEQ / 128) * (DIM / 128));
    gemm_bf16<true><<<gg, blk, 0, stream>>>(xb, wqb, bq, qraw, SEQ, DIM, DIM);
    gemm_bf16<true><<<gg, blk, 0, stream>>>(xb, wkb, bk, kraw, SEQ, DIM, DIM);
    gemm_bf16<true><<<gg, blk, 0, stream>>>(xb, wvb, bv, vraw, SEQ, DIM, DIM);

    norm_rope<<<dim3((3 * SEQ) / 8), blk, 0, stream>>>(
        qraw, kraw, vraw, gq, gk, fq, (uint32_t*)qbuf, (uint32_t*)kbuf, vbT);

    flash_attn<<<dim3(HEADS * (SEQ / 128)), blk, 0, stream>>>(qbuf, kbuf, vbT, obuf);

    gemm_bf16<false><<<gg, blk, 0, stream>>>(obuf, wob, bo, d_out, SEQ, DIM, DIM);
}
